// ConvNextBlock_19361712571151
// MI455X (gfx1250) — matrix operand order repaired and checked by reading, not compiled, not run on hardware
//
#include <hip/hip_runtime.h>
#include <stdint.h>

#define C_    256
#define L_    4096
#define B_    16
#define HID_  1024
#define EMB_  128
#define EPS_  1e-6f
#define LT    64        // L-tile per block
#define STRY  264       // yn LDS row stride (bf16 elems): 256 + 8 pad
#define STRH  136       // h  LDS row stride (bf16 elems): 128 + 8 pad
#define HCH   128       // hidden chunk (one 16-row tile per wave)

typedef __attribute__((ext_vector_type(16))) __bf16    v16bf;
typedef __attribute__((ext_vector_type(8)))  float     v8f;
typedef __attribute__((ext_vector_type(8)))  unsigned  v8u;

union Frag { v8u u; v16bf bf; };

__device__ inline uint16_t f2bf(float f) {
    union { float f; uint32_t u; } v; v.f = f;
    uint32_t u = v.u;
    return (uint16_t)((u + 0x7FFFu + ((u >> 16) & 1u)) >> 16);  // RNE
}

// ---------------------------------------------------------------------------
// Kernel 1: depthwise circular conv (K=7) + per-row sum/sumsq partials
// grid = B*C blocks, 256 threads
// ---------------------------------------------------------------------------
__global__ __launch_bounds__(256) void k_dwconv(
    const float* __restrict__ x, const float* __restrict__ dw_w,
    const float* __restrict__ dw_b, float* __restrict__ y,
    float* __restrict__ rsum, float* __restrict__ rsq)
{
    __shared__ float row[L_];
    __shared__ float red1[256], red2[256];
    int bc  = blockIdx.x;            // b*C + c
    int c   = bc & (C_ - 1);
    int tid = threadIdx.x;
    const float* xr = x + (size_t)bc * L_;
    #pragma unroll
    for (int i = 0; i < L_ / 256; ++i)
        row[tid + i * 256] = __builtin_nontemporal_load(&xr[tid + i * 256]);
    __syncthreads();

    float w0 = dw_w[c*7+0], w1 = dw_w[c*7+1], w2 = dw_w[c*7+2], w3 = dw_w[c*7+3],
          w4 = dw_w[c*7+4], w5 = dw_w[c*7+5], w6 = dw_w[c*7+6];
    float bias = dw_b[c];
    float s1 = 0.f, s2 = 0.f;
    float* yr = y + (size_t)bc * L_;
    #pragma unroll
    for (int i = 0; i < L_ / 256; ++i) {
        int l = tid + i * 256;
        float acc = bias;
        acc += w0 * row[(l - 3) & (L_ - 1)];
        acc += w1 * row[(l - 2) & (L_ - 1)];
        acc += w2 * row[(l - 1) & (L_ - 1)];
        acc += w3 * row[l];
        acc += w4 * row[(l + 1) & (L_ - 1)];
        acc += w5 * row[(l + 2) & (L_ - 1)];
        acc += w6 * row[(l + 3) & (L_ - 1)];
        yr[l] = acc;
        s1 += acc; s2 += acc * acc;
    }
    red1[tid] = s1; red2[tid] = s2;
    __syncthreads();
    for (int s = 128; s > 0; s >>= 1) {
        if (tid < s) { red1[tid] += red1[tid + s]; red2[tid] += red2[tid + s]; }
        __syncthreads();
    }
    if (tid == 0) { rsum[bc] = red1[0]; rsq[bc] = red2[0]; }
}

// ---------------------------------------------------------------------------
// Kernel 2: per-batch mean/inv_std + FiLM -> per-channel affine A,D
// grid = B blocks, 256 threads (thread t owns channel t)
// ---------------------------------------------------------------------------
__global__ __launch_bounds__(256) void k_finalize(
    const float* __restrict__ rsum, const float* __restrict__ rsq,
    const float* __restrict__ t,    const float* __restrict__ fw,
    const float* __restrict__ fb,   float* __restrict__ Ac,
    float* __restrict__ Dc)
{
    __shared__ float red1[256], red2[256];
    __shared__ float tsh[EMB_];
    __shared__ float stats[2];
    int b = blockIdx.x, tid = threadIdx.x;
    red1[tid] = rsum[b * C_ + tid];
    red2[tid] = rsq[b * C_ + tid];
    if (tid < EMB_) tsh[tid] = t[b * EMB_ + tid];
    __syncthreads();
    for (int s = 128; s > 0; s >>= 1) {
        if (tid < s) { red1[tid] += red1[tid + s]; red2[tid] += red2[tid + s]; }
        __syncthreads();
    }
    if (tid == 0) {
        float n   = (float)C_ * (float)L_;
        float mu  = red1[0] / n;
        float var = red2[0] / n - mu * mu;
        stats[0] = mu; stats[1] = rsqrtf(var + EPS_);
    }
    __syncthreads();
    float mu = stats[0], inv = stats[1];
    float sc = fb[tid], sh = fb[C_ + tid];
    const float* wr1 = fw + (size_t)tid * EMB_;
    const float* wr2 = fw + (size_t)(C_ + tid) * EMB_;
    #pragma unroll 8
    for (int k = 0; k < EMB_; ++k) { sc += tsh[k] * wr1[k]; sh += tsh[k] * wr2[k]; }
    float A = inv * (sc + 1.0f);
    Ac[b * C_ + tid] = A;
    Dc[b * C_ + tid] = sh - mu * A;
}

// ---------------------------------------------------------------------------
// Kernel 3: fp32 -> bf16 weight conversion (pw1 and pw2)
// ---------------------------------------------------------------------------
__global__ __launch_bounds__(256) void k_cvt(
    const float* __restrict__ a, const float* __restrict__ b,
    uint16_t* __restrict__ oa, uint16_t* __restrict__ ob)
{
    int i = blockIdx.x * 256 + threadIdx.x;
    oa[i] = f2bf(a[i]);
    ob[i] = f2bf(b[i]);
}

// ---------------------------------------------------------------------------
// Kernel 4: fused  yn=y*A+D (bf16)  ->  relu(pw1@yn+b1)  ->  pw2@h+b2
//           -> out = x + gamma * branch        (all WMMA bf16, f32 accum)
// grid = (L/LT, B), 256 threads = 8 waves; each wave owns one hidden
// 16-row tile in GEMM1 (weights read exactly once per block).
// ---------------------------------------------------------------------------
__global__ __launch_bounds__(256) void k_block(
    const float* __restrict__ y,  const float* __restrict__ Ac,
    const float* __restrict__ Dc, const uint16_t* __restrict__ w1,
    const uint16_t* __restrict__ w2, const float* __restrict__ b1,
    const float* __restrict__ b2,  const float* __restrict__ gamma,
    const float* __restrict__ x,   float* __restrict__ out)
{
    __shared__ __align__(16) uint16_t yn[LT * STRY];   // [l][c], bf16
    __shared__ __align__(16) uint16_t hl[LT * STRH];   // [l][h_local], bf16
    __shared__ float As[C_], Ds[C_], Bs[C_], Gs[C_];

    int b    = blockIdx.y;
    int l0   = blockIdx.x * LT;
    int tid  = threadIdx.x;
    int lane = tid & 31, wave = tid >> 5;
    int hlf  = lane >> 4, lm = lane & 15;

    As[tid] = Ac[b * C_ + tid];
    Ds[tid] = Dc[b * C_ + tid];
    Bs[tid] = b2[tid];
    Gs[tid] = gamma[tid];
    __syncthreads();

    // stage yn tile (256 channels x LT columns), l-major for B-fragment loads
    #pragma unroll 4
    for (int i = 0; i < LT; ++i) {
        int idx = tid + i * 256;
        int c = idx >> 6, l = idx & (LT - 1);
        float val = __builtin_nontemporal_load(
            &y[(((size_t)b * C_ + c) << 12) + l0 + l]);
        yn[l * STRY + c] = f2bf(val * As[c] + Ds[c]);
    }
    __syncthreads();

    v8f acc[2][4];
    #pragma unroll
    for (int si = 0; si < 2; ++si)
        #pragma unroll
        for (int ct = 0; ct < 4; ++ct)
            acc[si][ct] = (v8f){0.f,0.f,0.f,0.f,0.f,0.f,0.f,0.f};

    for (int hc = 0; hc < HID_ / HCH; ++hc) {
        int hb = hc * HCH;

        // ---- GEMM1: h[16 x LT] = pw1[16 x 256] @ yn[256 x LT] per wave ----
        v8f hf[4];
        #pragma unroll
        for (int ct = 0; ct < 4; ++ct)
            hf[ct] = (v8f){0.f,0.f,0.f,0.f,0.f,0.f,0.f,0.f};
        int hrow = hb + wave * 16 + lm;
        const uint16_t* w1r = w1 + (size_t)hrow * C_;
        if (hc < HID_ / HCH - 1)
            __builtin_prefetch((const void*)(w1r + (size_t)HCH * C_), 0, 3);
        #pragma unroll
        for (int kc = 0; kc < 8; ++kc) {
            Frag a;
            #pragma unroll
            for (int v = 0; v < 8; ++v) {
                int k0 = 2 * (v & 3) + 16 * (v >> 2) + 8 * hlf;
                a.u[v] = *(const unsigned*)(w1r + kc * 32 + k0);
            }
            #pragma unroll
            for (int ct = 0; ct < 4; ++ct) {
                Frag bb;
                #pragma unroll
                for (int v = 0; v < 8; ++v) {
                    int ks = 2 * (v & 3) + 16 * (v >> 2) + 8 * hlf;
                    bb.u[v] = *(const unsigned*)(&yn[(ct * 16 + lm) * STRY + kc * 32 + ks]);
                }
                hf[ct] = __builtin_amdgcn_wmma_f32_16x16x32_bf16(
                    false, a.bf, false, bb.bf, (short)0, hf[ct], false, false);
            }
        }
        // bias + relu, pack pairs, store h tile to LDS (l-major)
        float b1r[8];
        #pragma unroll
        for (int i = 0; i < 8; ++i) b1r[i] = b1[hb + wave * 16 + i + 8 * hlf];
        #pragma unroll
        for (int ct = 0; ct < 4; ++ct) {
            int lcol = ct * 16 + lm;
            #pragma unroll
            for (int i = 0; i < 8; i += 2) {
                float f0 = fmaxf(hf[ct][i]     + b1r[i],     0.f);
                float f1 = fmaxf(hf[ct][i + 1] + b1r[i + 1], 0.f);
                unsigned p = (unsigned)f2bf(f0) | ((unsigned)f2bf(f1) << 16);
                *(unsigned*)(&hl[lcol * STRH + wave * 16 + i + 8 * hlf]) = p;
            }
        }
        __syncthreads();

        // ---- GEMM2: branch[256 x LT] += pw2[256 x 128] @ h[128 x LT] ----
        #pragma unroll
        for (int kc2 = 0; kc2 < 4; ++kc2) {
            #pragma unroll
            for (int si = 0; si < 2; ++si) {
                int crow = (wave + si * 8) * 16 + lm;
                const uint16_t* w2r = w2 + (size_t)crow * HID_ + hb + kc2 * 32;
                if (kc2 == 0 && hc < HID_ / HCH - 1)
                    __builtin_prefetch((const void*)(w2r + HCH), 0, 3);
                Frag a;
                #pragma unroll
                for (int v = 0; v < 8; ++v) {
                    int k0 = 2 * (v & 3) + 16 * (v >> 2) + 8 * hlf;
                    a.u[v] = *(const unsigned*)(w2r + k0);
                }
                #pragma unroll
                for (int ct = 0; ct < 4; ++ct) {
                    Frag bb;
                    #pragma unroll
                    for (int v = 0; v < 8; ++v) {
                        int ks = 2 * (v & 3) + 16 * (v >> 2) + 8 * hlf;
                        bb.u[v] = *(const unsigned*)(&hl[(ct * 16 + lm) * STRH + kc2 * 32 + ks]);
                    }
                    acc[si][ct] = __builtin_amdgcn_wmma_f32_16x16x32_bf16(
                        false, a.bf, false, bb.bf, (short)0, acc[si][ct], false, false);
                }
            }
        }
        __syncthreads();
    }

    // epilogue: out = x + (acc + b2) * gamma   (streaming, non-temporal)
    #pragma unroll
    for (int si = 0; si < 2; ++si) {
        #pragma unroll
        for (int ct = 0; ct < 4; ++ct) {
            #pragma unroll
            for (int i = 0; i < 8; ++i) {
                int c = (wave + si * 8) * 16 + i + 8 * hlf;
                size_t idx = (((size_t)b * C_ + c) << 12) + l0 + ct * 16 + lm;
                float xv = __builtin_nontemporal_load(&x[idx]);
                __builtin_nontemporal_store(
                    xv + (acc[si][ct][i] + Bs[c]) * Gs[c], &out[idx]);
            }
        }
    }
}

// ---------------------------------------------------------------------------
extern "C" void kernel_launch(void* const* d_in, const int* in_sizes, int n_in,
                              void* d_out, int out_size, void* d_ws, size_t ws_size,
                              hipStream_t stream)
{
    const float* x     = (const float*)d_in[0];
    const float* t     = (const float*)d_in[1];
    const float* dw_w  = (const float*)d_in[2];
    const float* dw_b  = (const float*)d_in[3];
    const float* fw    = (const float*)d_in[4];
    const float* fb    = (const float*)d_in[5];
    const float* pw1_w = (const float*)d_in[6];
    const float* pw1_b = (const float*)d_in[7];
    const float* pw2_w = (const float*)d_in[8];
    const float* pw2_b = (const float*)d_in[9];
    const float* gamma = (const float*)d_in[10];
    float* out = (float*)d_out;

    char* ws = (char*)d_ws;
    size_t off = 0;
    float* y     = (float*)(ws + off); off += (size_t)B_ * C_ * L_ * 4;  // 64 MB
    float* rsum  = (float*)(ws + off); off += (size_t)B_ * C_ * 4;
    float* rsq   = (float*)(ws + off); off += (size_t)B_ * C_ * 4;
    float* Ac    = (float*)(ws + off); off += (size_t)B_ * C_ * 4;
    float* Dc    = (float*)(ws + off); off += (size_t)B_ * C_ * 4;
    uint16_t* w1 = (uint16_t*)(ws + off); off += (size_t)HID_ * C_ * 2;
    uint16_t* w2 = (uint16_t*)(ws + off); off += (size_t)HID_ * C_ * 2;

    k_dwconv  <<<dim3(B_ * C_),         dim3(256), 0, stream>>>(x, dw_w, dw_b, y, rsum, rsq);
    k_finalize<<<dim3(B_),              dim3(256), 0, stream>>>(rsum, rsq, t, fw, fb, Ac, Dc);
    k_cvt     <<<dim3(HID_ * C_ / 256), dim3(256), 0, stream>>>(pw1_w, pw2_w, w1, w2);
    k_block   <<<dim3(L_ / LT, B_),     dim3(256), 0, stream>>>(y, Ac, Dc, w1, w2,
                                                                pw1_b, pw2_b, gamma, x, out);
}
